// SlidingWindowAttention_78932908966332
// MI455X (gfx1250) — matrix operand order repaired and checked by reading, not compiled, not run on hardware
//
#include <hip/hip_runtime.h>
#include <hip/hip_bf16.h>

// ---------------------------------------------------------------------------
// CDNA5 (gfx1250) sliding-window attention, bf16 WMMA path.
// Shapes: B=2, S=2048, E=1024, H=16, D=64, WINDOW=256.
// ---------------------------------------------------------------------------

typedef __attribute__((ext_vector_type(16))) __bf16 bf16x16;
typedef __attribute__((ext_vector_type(8)))  __bf16 bf16x8;
typedef __attribute__((ext_vector_type(8)))  float  f32x8;

#define DEVINL __device__ __forceinline__

DEVINL bf16x16 cat16(bf16x8 lo, bf16x8 hi) {
  return __builtin_shufflevector(lo, hi, 0,1,2,3,4,5,6,7,8,9,10,11,12,13,14,15);
}

// A-fragment (16x32 bf16): lane holds row (lane&15).
// element e -> k = e + 16*(lane>>4)  => one contiguous 16-elem run.
DEVINL bf16x16 load_a(const __bf16* rowp, int half) {
  bf16x8 lo = *(const bf16x8*)(rowp + 16*half);
  bf16x8 hi = *(const bf16x8*)(rowp + 16*half + 8);
  return cat16(lo, hi);
}

// B-fragment (32x16 bf16): lane holds col (lane&15).
// element e -> k = e + 16*(lane>>4). 'colp' must already include +16*(lane>>4).
DEVINL bf16x16 load_b(const __bf16* colp) {
  bf16x8 lo = *(const bf16x8*)(colp);
  bf16x8 hi = *(const bf16x8*)(colp + 8);
  return cat16(lo, hi);
}

DEVINL f32x8 wmma_bf16(bf16x16 a, bf16x16 b, f32x8 c) {
  // 8 args: (neg_a, A, neg_b, B, c_mod, C, reuse_a, reuse_b)
  return __builtin_amdgcn_wmma_f32_16x16x32_bf16(false, a, false, b,
                                                 (short)0, c, false, false);
}

// 16-lane-group reductions (lanes {0..15} and {16..31}) via ds_swizzle xor.
template<int XM>
DEVINL float swz(float x) {
  int i = __builtin_amdgcn_ds_swizzle(__float_as_int(x), (XM << 10) | 0x1f);
  return __int_as_float(i);
}
DEVINL float redmax16(float x) {
  x = fmaxf(x, swz<1>(x)); x = fmaxf(x, swz<2>(x));
  x = fmaxf(x, swz<4>(x)); x = fmaxf(x, swz<8>(x));
  return x;
}
DEVINL float redsum16(float x) {
  x += swz<1>(x); x += swz<2>(x); x += swz<4>(x); x += swz<8>(x);
  return x;
}

// ---------------------------------------------------------------------------
// Elementwise f32 -> bf16 conversion
// ---------------------------------------------------------------------------
__global__ void cvt_bf16(const float* __restrict__ in, __bf16* __restrict__ out,
                         int n) {
  int i = blockIdx.x * blockDim.x + threadIdx.x;
  if (i < n) out[i] = (__bf16)in[i];
}

// ---------------------------------------------------------------------------
// GEMM: C[M=4096, N=1024] = A[4096,1024] * Bw[1024,1024]^T   (y = x @ W.T)
// Both operands are K-contiguous rows in memory.
// Block = 256 threads = 8 waves as 2(M) x 4(N); wave tile = 32x64.
// MODE 0: bf16 out, head-split  [B,H,S,D]
// MODE 1: bf16 out, transposed  [B,H,D,S]   (for V)
// MODE 2: f32  out, row-major   [M,N]
// ---------------------------------------------------------------------------
template<int MODE>
__global__ __launch_bounds__(256)
void gemm4096(const __bf16* __restrict__ A, const __bf16* __restrict__ Bw,
              void* __restrict__ outp) {
  constexpr int K = 1024;
  const int lane = threadIdx.x & 31;
  const int wave = threadIdx.x >> 5;
  const int half = lane >> 4, l16 = lane & 15;
  const int m0 = blockIdx.y * 64  + (wave >> 2) * 32;
  const int n0 = blockIdx.x * 256 + (wave & 3) * 64;

  f32x8 c[2][4] = {};

  const __bf16* a0 = A + (size_t)(m0 + l16) * K;
  const __bf16* a1 = a0 + (size_t)16 * K;
  const __bf16* b0 = Bw + (size_t)(n0 + l16) * K + 16 * half;

  for (int k0 = 0; k0 < K; k0 += 32) {
    if (k0 + 256 < K) __builtin_prefetch(a0 + k0 + 256, 0, 1);
    bf16x16 af0 = load_a(a0 + k0, half);
    bf16x16 af1 = load_a(a1 + k0, half);
#pragma unroll
    for (int j = 0; j < 4; ++j) {
      bf16x16 bf = load_b(b0 + (size_t)j * 16 * K + k0);
      c[0][j] = wmma_bf16(af0, bf, c[0][j]);
      c[1][j] = wmma_bf16(af1, bf, c[1][j]);
    }
  }

#pragma unroll
  for (int i = 0; i < 2; ++i)
#pragma unroll
    for (int j = 0; j < 4; ++j)
#pragma unroll
      for (int r = 0; r < 8; ++r) {
        int m = m0 + i * 16 + r + 8 * half;   // global row
        int n = n0 + j * 16 + l16;            // global col
        float v = c[i][j][r];
        if constexpr (MODE == 2) {
          ((float*)outp)[(size_t)m * 1024 + n] = v;
        } else {
          int b = m >> 11, s = m & 2047, h = n >> 6, d = n & 63;
          size_t idx;
          if constexpr (MODE == 0)
            idx = ((size_t)(b * 16 + h) * 2048 + s) * 64 + d;
          else
            idx = ((size_t)(b * 16 + h) * 64 + d) * 2048 + s;
          ((__bf16*)outp)[idx] = (__bf16)v;
        }
      }
}

// ---------------------------------------------------------------------------
// Sliding-window causal flash attention.
// Q,K: bf16 [B,H,S,D]; Vt: bf16 [B,H,D,S]; ctx out: bf16 [B,S,E].
// One wave per (b, h, 16-query tile); 4096 tasks = 512 blocks x 8 waves.
// Key loop in pairs of 16-key tiles (32 keys / iteration).
// ---------------------------------------------------------------------------
__global__ __launch_bounds__(256)
void attn_swa(const __bf16* __restrict__ Q, const __bf16* __restrict__ Kh,
              const __bf16* __restrict__ Vt, __bf16* __restrict__ ctx) {
  constexpr int S = 2048, D = 64;
  constexpr float SCALE = 0.125f;           // 1/sqrt(64)
  constexpr float L2E = 1.44269504088896f;  // log2(e)

  __shared__ alignas(16) __bf16 plds[8][16 * 32];   // 1 KB per wave

  const int lane = threadIdx.x & 31;
  const int wv = threadIdx.x >> 5;
  const int half = lane >> 4, l16 = lane & 15;
  const int task = blockIdx.x * 8 + wv;  // 0..4095
  const int qt = task & 127;             // 128 query tiles of 16
  const int bh = task >> 7;              // 0..31
  const int q0 = qt * 16;

  const __bf16* Qb = Q  + (size_t)bh * S * D;
  const __bf16* Kb = Kh + (size_t)bh * S * D;
  const __bf16* Vb = Vt + (size_t)bh * S * D;  // [D, S]

  // Q A-fragments for d-chunks [0,32) and [32,64)
  bf16x16 qa0, qa1;
  {
    const __bf16* qrow = Qb + (size_t)(q0 + l16) * D;
    qa0 = load_a(qrow, half);
    qa1 = load_a(qrow + 32, half);
  }

  f32x8 acc[4] = {};
  float rm[8], rl[8];
#pragma unroll
  for (int r = 0; r < 8; ++r) { rm[r] = -1e30f; rl[r] = 0.f; }

  int kstart = q0 - 256; if (kstart < 0) kstart = 0;
  kstart &= ~31;  // align to 32-key pairs; extra low keys get masked

  for (int kb = kstart; kb <= q0 + 15; kb += 32) {
    // ---- scores: two 16x16 f32 tiles over 32 keys ----
    f32x8 s[2] = {};
#pragma unroll
    for (int j = 0; j < 2; ++j) {
      int krow = kb + j * 16 + l16; if (krow > S - 1) krow = S - 1;  // safe
      const __bf16* kp = Kb + (size_t)krow * D + 16 * half;
      s[j] = wmma_bf16(qa0, load_b(kp), s[j]);
      s[j] = wmma_bf16(qa1, load_b(kp + 32), s[j]);
    }

    // ---- mask + online softmax update ----
    float p0v[8], p1v[8], alpha[8];
#pragma unroll
    for (int r = 0; r < 8; ++r) {
      int row = q0 + r + 8 * half;
      int k0i = kb + l16, k1i = kb + 16 + l16;
      bool ok0 = (k0i <= row) && (k0i >= row - 255);
      bool ok1 = (k1i <= row) && (k1i >= row - 255);
      float v0 = ok0 ? s[0][r] * SCALE : -1e30f;
      float v1 = ok1 ? s[1][r] * SCALE : -1e30f;
      float mx = redmax16(fmaxf(v0, v1));
      float mn = fmaxf(rm[r], mx);
      float al = exp2f((rm[r] - mn) * L2E);
      rm[r] = mn;
      float p0 = exp2f((v0 - mn) * L2E);
      float p1 = exp2f((v1 - mn) * L2E);
      rl[r] = rl[r] * al + redsum16(p0 + p1);
      p0v[r] = p0; p1v[r] = p1; alpha[r] = al;
    }
#pragma unroll
    for (int j = 0; j < 4; ++j)
#pragma unroll
      for (int r = 0; r < 8; ++r) acc[j][r] *= alpha[r];

    // ---- transpose P through per-wave LDS: C-layout -> A-layout ----
#pragma unroll
    for (int r = 0; r < 8; ++r) {
      plds[wv][(r + 8 * half) * 32 + l16]      = (__bf16)p0v[r];
      plds[wv][(r + 8 * half) * 32 + 16 + l16] = (__bf16)p1v[r];
    }
    asm volatile("" ::: "memory");  // DS ops are in-order within a wave
    bf16x16 pa;
    {
      const __bf16* pp = &plds[wv][l16 * 32];
      bf16x8 lo = *(const bf16x8*)(pp + 16 * half);
      bf16x8 hi = *(const bf16x8*)(pp + 16 * half + 8);
      pa = cat16(lo, hi);
    }

    // ---- ctx += P @ V  (V transposed layout => contiguous B frags) ----
    int ks = kb + 16 * half; if (ks > S - 16) ks = S - 16;  // safe
#pragma unroll
    for (int j = 0; j < 4; ++j) {
      const __bf16* vp = Vb + (size_t)(j * 16 + l16) * S + ks;
      acc[j] = wmma_bf16(pa, load_b(vp), acc[j]);
    }
  }

  // ---- epilogue: normalize and write ctx as [B,S,E] bf16 ----
  const int b = bh >> 4, h = bh & 15;
#pragma unroll
  for (int j = 0; j < 4; ++j)
#pragma unroll
    for (int r = 0; r < 8; ++r) {
      int srow = q0 + r + 8 * half;
      int col = h * 64 + j * 16 + l16;
      float o = acc[j][r] / rl[r];
      ctx[(size_t)(b * 2048 + srow) * 1024 + col] = (__bf16)o;
    }
}

// ---------------------------------------------------------------------------
// Launch
// ---------------------------------------------------------------------------
extern "C" void kernel_launch(void* const* d_in, const int* in_sizes, int n_in,
                              void* d_out, int out_size, void* d_ws,
                              size_t ws_size, hipStream_t stream) {
  (void)in_sizes; (void)n_in; (void)out_size; (void)ws_size;
  const float* x  = (const float*)d_in[0];
  const float* Wq = (const float*)d_in[1];
  const float* Wk = (const float*)d_in[2];
  const float* Wv = (const float*)d_in[3];
  const float* Wo = (const float*)d_in[4];
  float* out = (float*)d_out;

  char* ws = (char*)d_ws;
  const size_t MB = 1ull << 20;
  __bf16* xb  = (__bf16*)(ws + 0);        //  8 MiB  x bf16 [4096,1024]
  __bf16* wqb = (__bf16*)(ws + 8 * MB);   //  2 MiB
  __bf16* wkb = (__bf16*)(ws + 10 * MB);  //  2 MiB
  __bf16* wvb = (__bf16*)(ws + 12 * MB);  //  2 MiB
  __bf16* wob = (__bf16*)(ws + 14 * MB);  //  2 MiB
  __bf16* qh  = (__bf16*)(ws + 16 * MB);  //  8 MiB  Q [B,H,S,D]
  __bf16* kh  = (__bf16*)(ws + 24 * MB);  //  8 MiB  K [B,H,S,D]
  __bf16* vth = (__bf16*)(ws + 32 * MB);  //  8 MiB  V [B,H,D,S]
  __bf16* ctx = (__bf16*)(ws + 40 * MB);  //  8 MiB  ctx [B,S,E]

  const int NX = 4096 * 1024, NW = 1024 * 1024;
  cvt_bf16<<<(NX + 255) / 256, 256, 0, stream>>>(x, xb, NX);
  cvt_bf16<<<(NW + 255) / 256, 256, 0, stream>>>(Wq, wqb, NW);
  cvt_bf16<<<(NW + 255) / 256, 256, 0, stream>>>(Wk, wkb, NW);
  cvt_bf16<<<(NW + 255) / 256, 256, 0, stream>>>(Wv, wvb, NW);
  cvt_bf16<<<(NW + 255) / 256, 256, 0, stream>>>(Wo, wob, NW);

  dim3 g(4, 64);  // (N/256, M/64)
  gemm4096<0><<<g, 256, 0, stream>>>(xb, wqb, (void*)qh);
  gemm4096<0><<<g, 256, 0, stream>>>(xb, wkb, (void*)kh);
  gemm4096<1><<<g, 256, 0, stream>>>(xb, wvb, (void*)vth);

  attn_swa<<<512, 256, 0, stream>>>(qh, kh, vth, ctx);

  gemm4096<2><<<g, 256, 0, stream>>>(ctx, wob, (void*)out);
}
